// QuadpartiteHeteroGNN_7198365188425
// MI455X (gfx1250) — matrix operand order repaired and checked by reading, not compiled, not run on hardware
//
#include <hip/hip_runtime.h>
#include <math.h>

// ---------------------------------------------------------------------------
// CDNA5 / gfx1250 quadpartite GENConv GNN forward.
// Dense linears: v_wmma_f32_16x16x32_bf16 (fp32 accumulate), shapes templated
// so the K-loop fully unrolls and epilogues are straight-line code.
// Edge softmax aggregation: two L2-resident atomic passes (all per-relation
// tensors fit in MI455X's 192MB L2).
// ---------------------------------------------------------------------------

typedef __attribute__((ext_vector_type(16))) __bf16 v16bf;
typedef __attribute__((ext_vector_type(8)))  float  v8f;

#define GF_RELU  1
#define GF_ACCUM 2

#define GNN_EPS      1e-7f
#define GNN_BNSCALE  0.99999500003749968750f  /* 1/sqrt(1+1e-5) */

// ---------------------------------------------------------------------------
// GEMM: C[M,N] = epilogue(A[M,K] @ W[K,N] + bias [+ add]).
// Compile-time K,N,LDC,FLAGS,HASADD. M % 16 == 0 (50000, 100000, 1024).
// 256 threads = 8 waves; each wave owns one 16x16 tile.
//   tilesN = N/16 (4 or 8), rowTilesPerBlock = 8/tilesN.
// Weights are staged TRANSPOSED into LDS as bf16 with padded row stride
// (K+16 elems -> 32B-aligned rows), so a B fragment is one aligned 32-byte
// LDS vector load per WMMA.
// ---------------------------------------------------------------------------
template <int K, int N, int LDC, int FLAGS, bool HASADD>
__global__ __launch_bounds__(256) void gemm_k(
    const float* __restrict__ A, const float* __restrict__ W,
    const float* __restrict__ bias, const float* __restrict__ addsrc,
    float* __restrict__ C, int M, int coloff)
{
    constexpr int PK = K + 16;           // padded LDS row stride (elems)
    __shared__ __bf16 sWt[N * PK];       // transposed: sWt[n*PK + k]

    const int tid  = threadIdx.x;
    const int wave = tid >> 5;
    const int lane = tid & 31;
    const int r    = lane & 15;
    const int half = lane >> 4;

    // Stage W (K x N, row-major) transposed into LDS, coalesced global reads.
    #pragma unroll 4
    for (int i = tid; i < K * N; i += 256) {
        const int k = i / N;             // N is a power of two -> shifts
        const int n = i & (N - 1);
        sWt[n * PK + k] = (__bf16)W[i];
    }
    __syncthreads();

    constexpr int tilesN = N >> 4;
    constexpr int rtpb   = 8 / tilesN;
    const int colTile = wave % tilesN;
    const int rowTile = blockIdx.x * rtpb + wave / tilesN;
    const int row0    = rowTile << 4;
    if (row0 >= M) return;               // uniform per wave; EXEC stays full
    const int col0    = colTile << 4;
    const int col     = col0 + r;

    v8f acc = {};
    const float* Arow = A + (size_t)(row0 + r) * K;
    const v16bf* Bcol = reinterpret_cast<const v16bf*>(sWt + col * PK);

    #pragma unroll
    for (int kk = 0; kk < K; kk += 32) {
        // A fragment: 16x32 bf16: lane(r,half) holds
        // k = kk + half*16 + {0..15} (same order as B).
        const float* ap = Arow + kk + half * 16;
        float4 a0 = *reinterpret_cast<const float4*>(ap);
        float4 a1 = *reinterpret_cast<const float4*>(ap + 4);
        float4 a2 = *reinterpret_cast<const float4*>(ap + 8);
        float4 a3 = *reinterpret_cast<const float4*>(ap + 12);
        v16bf af;
        af[0]  = (__bf16)a0.x; af[1]  = (__bf16)a0.y;
        af[2]  = (__bf16)a0.z; af[3]  = (__bf16)a0.w;
        af[4]  = (__bf16)a1.x; af[5]  = (__bf16)a1.y;
        af[6]  = (__bf16)a1.z; af[7]  = (__bf16)a1.w;
        af[8]  = (__bf16)a2.x; af[9]  = (__bf16)a2.y;
        af[10] = (__bf16)a2.z; af[11] = (__bf16)a2.w;
        af[12] = (__bf16)a3.x; af[13] = (__bf16)a3.y;
        af[14] = (__bf16)a3.z; af[15] = (__bf16)a3.w;

        // B fragment: 32x16 bf16; lane r = column, element e -> K = kk +
        // half*16 + e. With the transposed LDS layout this is one aligned
        // 32-byte vector load: sWt[col*PK + kk + half*16 .. +15].
        const v16bf bfrag = *reinterpret_cast<const v16bf*>(
            reinterpret_cast<const __bf16*>(Bcol) + kk + half * 16);

        acc = __builtin_amdgcn_wmma_f32_16x16x32_bf16(
            false, af, false, bfrag, (short)0, acc, false, false);
    }

    // Epilogue (compile-time specialized, straight-line).
    // C/D layout: lane(r,half) -> col = col0+r, rows m = row0 + half*8 + v.
    const float bv   = bias[col];
    const int  mbase = row0 + half * 8;
    float*       cp  = C + (size_t)mbase * LDC + coloff + col;
    const float* ap2 = HASADD ? (addsrc + (size_t)mbase * N + col) : nullptr;
    #pragma unroll
    for (int v = 0; v < 8; ++v) {
        float val = acc[v] + bv;
        if constexpr (HASADD)            val += ap2[v * N];
        if constexpr (FLAGS & GF_RELU)   val = fmaxf(val, 0.0f);
        if constexpr (FLAGS & GF_ACCUM)  cp[v * LDC] += val;
        else                             cp[v * LDC]  = val;
    }
}

template <int K, int N, int LDC, int FLAGS, bool HASADD>
static inline void launch_gemm(const float* A, const float* W, const float* b,
                               const float* add, float* C, int M, int coloff,
                               hipStream_t stream)
{
    constexpr int tilesN = N >> 4;
    constexpr int rtpb   = 8 / tilesN;
    const int blocks = (M / 16 + rtpb - 1) / rtpb;
    gemm_k<K, N, LDC, FLAGS, HASADD><<<blocks, 256, 0, stream>>>(
        A, W, b, add, C, M, coloff);
}

// ---------------------------------------------------------------------------
// Encoder layer 1: y[n,j] = relu( (x[n,:4] @ W1[:,j] + b1[j]) * BN_SCALE )
// ---------------------------------------------------------------------------
__global__ __launch_bounds__(256) void enc_l1_k(
    const float* __restrict__ x, const float* __restrict__ W1,
    const float* __restrict__ b1, float* __restrict__ y, int Nn)
{
    int g = blockIdx.x * blockDim.x + threadIdx.x;
    if (g >= Nn * 64) return;
    int n = g >> 6, j = g & 63;
    const float* xr = x + (size_t)n * 4;
    float s = b1[j];
    s = fmaf(xr[0], W1[0 * 64 + j], s);
    s = fmaf(xr[1], W1[1 * 64 + j], s);
    s = fmaf(xr[2], W1[2 * 64 + j], s);
    s = fmaf(xr[3], W1[3 * 64 + j], s);
    y[g] = fmaxf(s * GNN_BNSCALE, 0.0f);
}

// ---------------------------------------------------------------------------
// Edge pass 1: msg = relu(h[src] + attr*We + Be) + eps ; atomic segment-max.
// msg > 0 and mx is zero-initialized, so signed-int atomicMax on the bit
// pattern matches float ordering. One thread per (edge, 4 columns).
// ---------------------------------------------------------------------------
__global__ __launch_bounds__(256) void edge_pass1_k(
    const int* __restrict__ src, const int* __restrict__ dst,
    const float* __restrict__ attr, const float* __restrict__ h,
    const float* __restrict__ We, const float* __restrict__ Be,
    float* __restrict__ mx, int E)
{
    int g = blockIdx.x * blockDim.x + threadIdx.x;
    if (g >= E * 16) return;
    int e = g >> 4, q = (g & 15) << 2;
    int s = src[e], d = dst[e];
    float a = attr[e];
    float4 h4 = *reinterpret_cast<const float4*>(h + (size_t)s * 64 + q);
    float4 w4 = *reinterpret_cast<const float4*>(We + q);
    float4 b4 = *reinterpret_cast<const float4*>(Be + q);
    float m0 = fmaxf(fmaf(a, w4.x, h4.x) + b4.x, 0.0f) + GNN_EPS;
    float m1 = fmaxf(fmaf(a, w4.y, h4.y) + b4.y, 0.0f) + GNN_EPS;
    float m2 = fmaxf(fmaf(a, w4.z, h4.z) + b4.z, 0.0f) + GNN_EPS;
    float m3 = fmaxf(fmaf(a, w4.w, h4.w) + b4.w, 0.0f) + GNN_EPS;
    int* mp = (int*)(mx + (size_t)d * 64 + q);
    atomicMax(mp + 0, __float_as_int(m0));
    atomicMax(mp + 1, __float_as_int(m1));
    atomicMax(mp + 2, __float_as_int(m2));
    atomicMax(mp + 3, __float_as_int(m3));
}

// ---------------------------------------------------------------------------
// Edge pass 2: ex = exp(msg - mx[dst]); denom += ex; sumw += msg*ex.
// ---------------------------------------------------------------------------
__global__ __launch_bounds__(256) void edge_pass2_k(
    const int* __restrict__ src, const int* __restrict__ dst,
    const float* __restrict__ attr, const float* __restrict__ h,
    const float* __restrict__ We, const float* __restrict__ Be,
    const float* __restrict__ mx, float* __restrict__ denom,
    float* __restrict__ sumw, int E)
{
    int g = blockIdx.x * blockDim.x + threadIdx.x;
    if (g >= E * 16) return;
    int e = g >> 4, q = (g & 15) << 2;
    int s = src[e], d = dst[e];
    float a = attr[e];
    float4 h4 = *reinterpret_cast<const float4*>(h + (size_t)s * 64 + q);
    float4 w4 = *reinterpret_cast<const float4*>(We + q);
    float4 b4 = *reinterpret_cast<const float4*>(Be + q);
    float4 x4 = *reinterpret_cast<const float4*>(mx + (size_t)d * 64 + q);
    float m0 = fmaxf(fmaf(a, w4.x, h4.x) + b4.x, 0.0f) + GNN_EPS;
    float m1 = fmaxf(fmaf(a, w4.y, h4.y) + b4.y, 0.0f) + GNN_EPS;
    float m2 = fmaxf(fmaf(a, w4.z, h4.z) + b4.z, 0.0f) + GNN_EPS;
    float m3 = fmaxf(fmaf(a, w4.w, h4.w) + b4.w, 0.0f) + GNN_EPS;
    float e0 = __expf(m0 - x4.x);
    float e1 = __expf(m1 - x4.y);
    float e2 = __expf(m2 - x4.z);
    float e3 = __expf(m3 - x4.w);
    float* dp = denom + (size_t)d * 64 + q;
    float* sp = sumw  + (size_t)d * 64 + q;
    atomicAdd(dp + 0, e0); atomicAdd(dp + 1, e1);
    atomicAdd(dp + 2, e2); atomicAdd(dp + 3, e3);
    atomicAdd(sp + 0, m0 * e0); atomicAdd(sp + 1, m1 * e1);
    atomicAdd(sp + 2, m2 * e2); atomicAdd(sp + 3, m3 * e3);
}

// agg = sumw / max(denom, 1e-16)   (in place on sumw)
__global__ __launch_bounds__(256) void finish_agg_k(
    float* __restrict__ sumw, const float* __restrict__ denom, int total)
{
    int g = blockIdx.x * blockDim.x + threadIdx.x;
    if (g >= total) return;
    sumw[g] = sumw[g] / fmaxf(denom[g], 1e-16f);
}

// x = (relu(h2) + x) * 0.5, over the whole contiguous node-feature region
__global__ __launch_bounds__(256) void update_x_k(
    const float* __restrict__ h2, float* __restrict__ x, int total)
{
    int g = blockIdx.x * blockDim.x + threadIdx.x;
    if (g >= total) return;
    x[g] = (fmaxf(h2[g], 0.0f) + x[g]) * 0.5f;
}

// Prediction head tail: out[n*2+step] = [relu]( dot(t0[n,:64], W2) + b2 )
__global__ __launch_bounds__(256) void pred_final_k(
    const float* __restrict__ t0, const float* __restrict__ W2,
    const float* __restrict__ b2, float* __restrict__ out,
    int Nn, int dorelu, int step)
{
    int n = blockIdx.x * blockDim.x + threadIdx.x;
    if (n >= Nn) return;
    const float* tr = t0 + (size_t)n * 64;
    float s = b2[0];
    #pragma unroll
    for (int j = 0; j < 64; ++j) s = fmaf(tr[j], W2[j], s);
    if (dorelu) s = fmaxf(s, 0.0f);
    out[(size_t)n * 2 + step] = s;
}

// ---------------------------------------------------------------------------
// Host orchestration
// ---------------------------------------------------------------------------
extern "C" void kernel_launch(void* const* d_in, const int* in_sizes, int n_in,
                              void* d_out, int out_size, void* d_ws, size_t ws_size,
                              hipStream_t stream)
{
    (void)in_sizes; (void)n_in; (void)out_size; (void)ws_size;

    // ---- input leaf index map (setup_inputs() dict insertion order) ----
    int idx = 0;
    int encW1[4], encB1[4], encW2[4], encB2[4];
    for (int t = 0; t < 4; ++t) {               // cons, econs, vals, obj
        encW1[t] = idx++; encB1[t] = idx++;
        encW2[t] = idx++; encB2[t] = idx++;
    }
    // convs[layer][rel][srcW,srcB,edgeW,edgeB,dstW,dstB,m1W,m1B,m2W,m2B]
    int cv[2][10][10];
    for (int l = 0; l < 2; ++l)
        for (int r = 0; r < 10; ++r)
            for (int p = 0; p < 10; ++p) cv[l][r][p] = idx++;
    int prW1[3], prB1[3], prW2[3], prB2[3];     // vals, cons, econs
    for (int t = 0; t < 3; ++t) {
        prW1[t] = idx++; prB1[t] = idx++;
        prW2[t] = idx++; prB2[t] = idx++;
    }
    int xin[4]; for (int t = 0; t < 4; ++t) xin[t] = idx++;   // cons,econs,vals,obj
    int ei[10]; for (int r = 0; r < 10; ++r) ei[r] = idx++;   // [2,E] int32
    int ea[10]; for (int r = 0; r < 10; ++r) ea[r] = idx++;   // [E,1] f32

    auto F = [&](int i) { return (const float*)d_in[i]; };

    // ---- sizes / relation table (type ids: 0 cons, 1 econs, 2 vals, 3 obj)
    const int Nt[4] = { 50000, 50000, 100000, 1024 };
    struct Rel { int s, d, E, h2t, coloff; };
    const Rel rels[10] = {
        {0, 2, 1000000, 2,  0},  // c2v
        {2, 0, 1000000, 0,  0},  // v2c
        {1, 2, 1000000, 2,  0},  // e2v
        {2, 1, 1000000, 1,  0},  // v2e
        {2, 3,  100000, 3, 64},  // v2o
        {3, 2,  100000, 2, 64},  // o2v
        {0, 3,   50000, 3,  0},  // c2o
        {3, 0,   50000, 0, 64},  // o2c
        {1, 3,   50000, 3,  0},  // e2o
        {3, 1,   50000, 1, 64},  // o2e
    };

    // ---- workspace layout (floats) ----
    float* ws = (float*)d_ws;
    size_t off = 0;
    float* Xp[4];
    for (int t = 0; t < 4; ++t) { Xp[t] = ws + off; off += (size_t)Nt[t] * 128; }
    const size_t xElems = off;                       // contiguous X region
    float* H2p[4];
    const size_t h2off = off;
    for (int t = 0; t < 4; ++t) { H2p[t] = ws + off; off += (size_t)Nt[t] * 128; }
    float* h2base = ws + h2off;
    const size_t h2Elems = off - h2off;              // == xElems
    float* hsrc  = ws + off; off += (size_t)100000 * 64;
    float* mxb   = ws + off; off += (size_t)100000 * 64;
    float* denom = ws + off; off += (size_t)100000 * 64;
    float* sumw  = ws + off; off += (size_t)100000 * 64;
    float* t0b   = ws + off; off += (size_t)100000 * 64;
    float* t1b   = ws + off; off += (size_t)100000 * 128;

    auto cdiv = [](long a, long b) { return (int)((a + b - 1) / b); };

    // ---------------- encoders: x[t] = l2( relu(l1(xin)*BN) ) --------------
    for (int t = 0; t < 4; ++t) {
        enc_l1_k<<<cdiv((long)Nt[t] * 64, 256), 256, 0, stream>>>(
            F(xin[t]), F(encW1[t]), F(encB1[t]), hsrc, Nt[t]);
        launch_gemm<64, 128, 128, 0, false>(
            hsrc, F(encW2[t]), F(encB2[t]), nullptr, Xp[t], Nt[t], 0, stream);
    }

    float* outp = (float*)d_out;
    const size_t outBase[3] = { 0, 200000, 300000 };  // vals, cons, econs
    const int    predType[3] = { 2, 0, 1 };
    const int    predRelu[3] = { 1, 0, 0 };

    // ---------------- IPM steps x layers -----------------------------------
    for (int step = 0; step < 2; ++step) {
        for (int layer = 0; layer < 2; ++layer) {
            hipMemsetAsync(h2base, 0, h2Elems * sizeof(float), stream);

            for (int r = 0; r < 10; ++r) {
                const Rel R = rels[r];
                const int* srcp = (const int*)d_in[ei[r]];
                const int* dstp = srcp + R.E;
                const float* attr = F(ea[r]);
                const int* P = cv[layer][r];

                // h = lin_src(x_src)            [Ns,128] @ [128,64]
                launch_gemm<128, 64, 64, 0, false>(
                    Xp[R.s], F(P[0]), F(P[1]), nullptr, hsrc,
                    Nt[R.s], 0, stream);

                const size_t dbytes = (size_t)Nt[R.d] * 64 * sizeof(float);
                hipMemsetAsync(mxb,   0, dbytes, stream);
                hipMemsetAsync(denom, 0, dbytes, stream);
                hipMemsetAsync(sumw,  0, dbytes, stream);

                const int eThreads = R.E * 16;
                edge_pass1_k<<<cdiv(eThreads, 256), 256, 0, stream>>>(
                    srcp, dstp, attr, hsrc, F(P[2]), F(P[3]), mxb, R.E);
                edge_pass2_k<<<cdiv(eThreads, 256), 256, 0, stream>>>(
                    srcp, dstp, attr, hsrc, F(P[2]), F(P[3]),
                    mxb, denom, sumw, R.E);
                finish_agg_k<<<cdiv((long)Nt[R.d] * 64, 256), 256, 0, stream>>>(
                    sumw, denom, Nt[R.d] * 64);

                // t0 = agg + lin_dst(x_dst)     [Nd,128] @ [128,64] + sumw
                launch_gemm<128, 64, 64, 0, true>(
                    Xp[R.d], F(P[4]), F(P[5]), sumw, t0b,
                    Nt[R.d], 0, stream);
                // t1 = relu(m1(t0))             [Nd,64] @ [64,128]
                launch_gemm<64, 128, 128, GF_RELU, false>(
                    t0b, F(P[6]), F(P[7]), nullptr, t1b,
                    Nt[R.d], 0, stream);
                // h2[dst][:, coloff:+64] += m2(t1)   [Nd,128] @ [128,64]
                launch_gemm<128, 64, 128, GF_ACCUM, false>(
                    t1b, F(P[8]), F(P[9]), nullptr, H2p[R.h2t],
                    Nt[R.d], R.coloff, stream);
            }

            // x = (relu(h2) + x) * 0.5 over the whole node region
            update_x_k<<<cdiv((long)xElems, 256), 256, 0, stream>>>(
                h2base, ws, (int)xElems);

            if (layer == 1) {  // end of IPM step: prediction heads on h2
                for (int j = 0; j < 3; ++j) {
                    const int t = predType[j];
                    launch_gemm<128, 64, 64, GF_RELU, false>(
                        H2p[t], F(prW1[j]), F(prB1[j]), nullptr, t0b,
                        Nt[t], 0, stream);
                    pred_final_k<<<cdiv(Nt[t], 256), 256, 0, stream>>>(
                        t0b, F(prW2[j]), F(prB2[j]), outp + outBase[j],
                        Nt[t], predRelu[j], step);
                }
            }
        }
    }
}
